// Mu_Infor_Spatial_23605140259218
// MI455X (gfx1250) — matrix operand order repaired and checked by reading, not compiled, not run on hardware
//
#include <hip/hip_runtime.h>
#include <math.h>

typedef __attribute__((ext_vector_type(16))) __bf16 v16bf;
typedef __attribute__((ext_vector_type(8)))  __bf16 v8bf;
typedef __attribute__((ext_vector_type(8)))  float  v8f;

#define BATCH 16
#define CDIM  256
#define HDIM  96
#define WDIM  96
#define HWDIM 9216      // 96*96
#define HSW   48
#define HWS   2304      // 48*48
#define KDIM  9216

static __device__ __forceinline__ int clampi(int v, int lo, int hi) {
    return v < lo ? lo : (v > hi ? hi : v);
}

// ---------------------------------------------------------------------------
// Kernel 1: per-(b,c) mean and centered L2 norm of f_p
// ---------------------------------------------------------------------------
__global__ __launch_bounds__(256)
void stats_p_kernel(const float* __restrict__ fp,
                    float* __restrict__ meanp, float* __restrict__ pn)
{
    __shared__ float redA[256];
    __shared__ float redB[256];
    const int bc = blockIdx.x;
    const int t  = threadIdx.x;
    const float* row = fp + (size_t)bc * HWDIM;
    float s = 0.f, s2 = 0.f;
    for (int r = 0; r < 36; ++r) {
        float v = row[t + 256 * r];
        s += v; s2 += v * v;
    }
    redA[t] = s; redB[t] = s2; __syncthreads();
    for (int st = 128; st > 0; st >>= 1) {
        if (t < st) { redA[t] += redA[t + st]; redB[t] += redB[t + st]; }
        __syncthreads();
    }
    if (t == 0) {
        float mean = redA[0] * (1.0f / HWDIM);
        float ssd  = fmaxf(redB[0] - (float)HWDIM * mean * mean, 0.0f);
        meanp[bc] = mean;
        pn[bc]    = sqrtf(ssd);
    }
}

// ---------------------------------------------------------------------------
// Kernel 2: write centered p_d as bf16
// ---------------------------------------------------------------------------
__global__ __launch_bounds__(256)
void center_p_kernel(const float* __restrict__ fp,
                     const float* __restrict__ meanp,
                     __bf16* __restrict__ pd)
{
    const int bc = blockIdx.x;
    const int t  = threadIdx.x;
    const float mean = meanp[bc];
    const float* row = fp + (size_t)bc * HWDIM;
    __bf16* orow = pd + (size_t)bc * HWDIM;
    for (int r = 0; r < 36; ++r) {
        int e = t + 256 * r;
        orow[e] = (__bf16)(row[e] - mean);
    }
}

// ---------------------------------------------------------------------------
// Kernel 3: bilinear upsample f_ms 48->96 (half-pixel, clamped), center,
//           write bf16 ms_d + norms
// ---------------------------------------------------------------------------
__global__ __launch_bounds__(256)
void resize_ms_kernel(const float* __restrict__ fms,
                      __bf16* __restrict__ msd, float* __restrict__ msn)
{
    __shared__ float tile[HWS];
    __shared__ float redA[256];
    __shared__ float redB[256];
    const int bc = blockIdx.x;
    const int t  = threadIdx.x;
    const float* src = fms + (size_t)bc * HWS;
    for (int r = 0; r < 9; ++r) tile[t + 256 * r] = src[t + 256 * r];
    __syncthreads();

    float vals[36];
    float s = 0.f, s2 = 0.f;
    for (int r = 0; r < 36; ++r) {
        int o  = t + 256 * r;
        int oh = o / WDIM;
        int ow = o - oh * WDIM;
        float fy = 0.5f * (float)oh - 0.25f;
        int   y0 = (int)floorf(fy);
        float wy = fy - (float)y0;
        int y0c = clampi(y0,     0, HSW - 1);
        int y1c = clampi(y0 + 1, 0, HSW - 1);
        float fx = 0.5f * (float)ow - 0.25f;
        int   x0 = (int)floorf(fx);
        float wx = fx - (float)x0;
        int x0c = clampi(x0,     0, HSW - 1);
        int x1c = clampi(x0 + 1, 0, HSW - 1);
        float v00 = tile[y0c * HSW + x0c], v01 = tile[y0c * HSW + x1c];
        float v10 = tile[y1c * HSW + x0c], v11 = tile[y1c * HSW + x1c];
        float v = (1.f - wy) * ((1.f - wx) * v00 + wx * v01)
                +        wy  * ((1.f - wx) * v10 + wx * v11);
        vals[r] = v; s += v; s2 += v * v;
    }
    redA[t] = s; redB[t] = s2; __syncthreads();
    for (int st = 128; st > 0; st >>= 1) {
        if (t < st) { redA[t] += redA[t + st]; redB[t] += redB[t + st]; }
        __syncthreads();
    }
    const float mean = redA[0] * (1.0f / HWDIM);
    __bf16* orow = msd + (size_t)bc * HWDIM;
    for (int r = 0; r < 36; ++r)
        orow[t + 256 * r] = (__bf16)(vals[r] - mean);
    if (t == 0) {
        float ssd = fmaxf(redB[0] - (float)HWDIM * mean * mean, 0.0f);
        msn[bc] = sqrtf(ssd);
    }
}

// ---------------------------------------------------------------------------
// Kernel 4: batched Gram GEMM with WMMA bf16 -> f32, fused 100/(ms_n*p_n)
//   block = 128 threads (4 waves), 64x64 output tile, K chunk = 32
// ---------------------------------------------------------------------------
__global__ __launch_bounds__(128)
void gemm_score_kernel(const __bf16* __restrict__ msd,
                       const __bf16* __restrict__ pd,
                       const float* __restrict__ msn,
                       const float* __restrict__ pn,
                       float* __restrict__ sm)
{
    __shared__ __attribute__((aligned(16))) __bf16 As[64][40];  // pad: row = 80B
    __shared__ __attribute__((aligned(16))) __bf16 Bs[64][40];

    const int blk  = blockIdx.x;
    const int b    = blk >> 4;
    const int tile = blk & 15;
    const int i0   = (tile >> 2) * 64;
    const int j0   = (tile & 3)  * 64;

    const int t    = threadIdx.x;
    const int lane = t & 31;
    const int wave = t >> 5;
    const int wi   = (wave >> 1) * 32;
    const int wj   = (wave & 1)  * 32;
    const int lm   = lane & 15;
    const int hi   = lane >> 4;

    const __bf16* Abase = msd + ((size_t)b * CDIM + i0) * KDIM;
    const __bf16* Bbase = pd  + ((size_t)b * CDIM + j0) * KDIM;

    const int lr = t >> 1;          // 0..63  row loaded by this thread
    const int lh = (t & 1) * 16;    // 0 or 16 (16 bf16 per thread per tile)

    v8f acc00 = {}, acc01 = {}, acc10 = {}, acc11 = {};

    for (int kk = 0; kk < KDIM; kk += 32) {
        __syncthreads();
        const uint4* ga = (const uint4*)(Abase + (size_t)lr * KDIM + kk + lh);
        const uint4* gb = (const uint4*)(Bbase + (size_t)lr * KDIM + kk + lh);
        uint4 a0 = ga[0], a1 = ga[1];
        uint4 b0 = gb[0], b1 = gb[1];
        *(uint4*)&As[lr][lh]     = a0;
        *(uint4*)&As[lr][lh + 8] = a1;
        *(uint4*)&Bs[lr][lh]     = b0;
        *(uint4*)&Bs[lr][lh + 8] = b1;
        if (kk + 32 < KDIM) {
            __builtin_prefetch((const void*)(Abase + (size_t)lr * KDIM + kk + 32 + lh), 0, 0);
            __builtin_prefetch((const void*)(Bbase + (size_t)lr * KDIM + kk + 32 + lh), 0, 0);
        }
        __syncthreads();

        // A fragments: 16-bit A layout -> lane<16 holds K{0..7,16..23} of row lm
        union { v16bf v; v8bf h[2]; } fa0, fa1, fb0, fb1;
        const int klo = hi * 8;
        {
            int r = wi + lm;
            fa0.h[0] = *(const v8bf*)&As[r][klo];
            fa0.h[1] = *(const v8bf*)&As[r][klo + 16];
            r = wi + 16 + lm;
            fa1.h[0] = *(const v8bf*)&As[r][klo];
            fa1.h[1] = *(const v8bf*)&As[r][klo + 16];
        }
        // B fragments: lane<16 holds K{0..7,16..23} of column lm, lane>=16 K{8..15,24..31}
        const int kb = hi * 8;
        {
            int c = wj + lm;
            fb0.h[0] = *(const v8bf*)&Bs[c][kb];
            fb0.h[1] = *(const v8bf*)&Bs[c][kb + 16];
            c = wj + 16 + lm;
            fb1.h[0] = *(const v8bf*)&Bs[c][kb];
            fb1.h[1] = *(const v8bf*)&Bs[c][kb + 16];
        }
        acc00 = __builtin_amdgcn_wmma_f32_16x16x32_bf16(false, fa0.v, false, fb0.v, (short)0, acc00, false, false);
        acc01 = __builtin_amdgcn_wmma_f32_16x16x32_bf16(false, fa0.v, false, fb1.v, (short)0, acc01, false, false);
        acc10 = __builtin_amdgcn_wmma_f32_16x16x32_bf16(false, fa1.v, false, fb0.v, (short)0, acc10, false, false);
        acc11 = __builtin_amdgcn_wmma_f32_16x16x32_bf16(false, fa1.v, false, fb1.v, (short)0, acc11, false, false);
    }

    // Epilogue: s_m[b,i,j] = num * 100 / (ms_n[b,i] * p_n[b,i])
    const float* msnb = msn + b * CDIM;
    const float* pnb  = pn  + b * CDIM;
    float* smb = sm + (size_t)b * CDIM * CDIM;
    const int jc0 = j0 + wj + lm;
    const int jc1 = j0 + wj + 16 + lm;
    for (int g = 0; g < 8; ++g) {
        int ir0 = i0 + wi + g + 8 * hi;   // C/D layout: lanes>=16 hold M=g+8
        int ir1 = ir0 + 16;
        float s0 = 100.0f / (msnb[ir0] * pnb[ir0]);
        float s1 = 100.0f / (msnb[ir1] * pnb[ir1]);
        smb[(size_t)ir0 * CDIM + jc0] = acc00[g] * s0;
        smb[(size_t)ir0 * CDIM + jc1] = acc01[g] * s0;
        smb[(size_t)ir1 * CDIM + jc0] = acc10[g] * s1;
        smb[(size_t)ir1 * CDIM + jc1] = acc11[g] * s1;
    }
}

// ---------------------------------------------------------------------------
// Kernel 5: group selection — argmax, softmax, deterministic scatter,
//           stable descending sort (bitonic with index tie-break)
// ---------------------------------------------------------------------------
__global__ __launch_bounds__(256)
void group_select_kernel(const float* __restrict__ sm,
                         float* __restrict__ scores,
                         int* __restrict__ vals,
                         int* __restrict__ cnt)
{
    __shared__ float red[256];
    __shared__ float mvArr[256];   // reused later as gs array
    __shared__ int   idxArr[256];  // reused later as present array
    __shared__ float skey[256];
    __shared__ int   spos[256];

    const int b = blockIdx.x;
    const int t = threadIdx.x;

    // row max / first argmax over j
    const float* row = sm + ((size_t)b * CDIM + t) * CDIM;
    float m = row[0]; int idx = 0;
    for (int j = 1; j < CDIM; ++j) {
        float v = row[j];
        if (v > m) { m = v; idx = j; }
    }

    // softmax over channels of the row-max values
    red[t] = m; __syncthreads();
    for (int st = 128; st > 0; st >>= 1) {
        if (t < st) red[t] = fmaxf(red[t], red[t + st]);
        __syncthreads();
    }
    float M = red[0]; __syncthreads();
    float e = expf(m - M);
    red[t] = e; __syncthreads();
    for (int st = 128; st > 0; st >>= 1) {
        if (t < st) red[t] += red[t + st];
        __syncthreads();
    }
    float mv = e / red[0]; __syncthreads();

    // deterministic scatter: thread j scans all i in ascending order
    mvArr[t] = mv; idxArr[t] = idx;
    __syncthreads();
    float gs = 0.0f; int pres = 0;
    for (int i = 0; i < CDIM; ++i) {
        if (idxArr[i] == t) { gs += mvArr[i]; pres = 1; }
    }
    __syncthreads();
    mvArr[t]  = gs;    // gs array
    idxArr[t] = pres;  // present array
    red[t] = (float)pres;
    __syncthreads();
    for (int st = 128; st > 0; st >>= 1) {
        if (t < st) red[t] += red[t + st];
        __syncthreads();
    }
    if (t == 0) cnt[b] = (int)(red[0] + 0.5f);

    // reverse (desc_vals indexing), mask non-present to -inf
    const int p = 255 - t;
    skey[t] = idxArr[p] ? mvArr[p] : -__builtin_inff();
    spos[t] = t;

    // bitonic sort: key descending, position ascending on ties (== stable argsort)
    for (int k = 2; k <= 256; k <<= 1) {
        for (int j = k >> 1; j > 0; j >>= 1) {
            __syncthreads();
            int ixj = t ^ j;
            if (ixj > t) {
                float ka = skey[t], kb2 = skey[ixj];
                int   pa = spos[t], pb  = spos[ixj];
                bool beforeT = (ka > kb2) || (ka == kb2 && pa < pb);
                bool dir = ((t & k) == 0);
                if (dir ? !beforeT : beforeT) {
                    skey[t] = kb2; skey[ixj] = ka;
                    spos[t] = pb;  spos[ixj] = pa;
                }
            }
        }
    }
    __syncthreads();
    scores[b * CDIM + t] = skey[t];
    vals[b * CDIM + t]   = 255 - spos[t];   // desc_vals[order]
}

// ---------------------------------------------------------------------------
// Kernel 6: min_k = ceil(min_b cnt_b / 2) + softmax of selected scores
// ---------------------------------------------------------------------------
__global__ __launch_bounds__(256)
void mink_weights_kernel(const int* __restrict__ cnt,
                         const float* __restrict__ scores,
                         int* __restrict__ minkp,
                         float* __restrict__ w)
{
    __shared__ float red[256];
    __shared__ int   mk;
    const int t = threadIdx.x;
    if (t == 0) {
        int m = cnt[0];
        for (int b = 1; b < BATCH; ++b) m = min(m, cnt[b]);
        int mink = (m + 1) >> 1;
        minkp[0] = mink;
        mk = mink;
    }
    __syncthreads();
    const int mink = mk;
    for (int b = 0; b < BATCH; ++b) {
        float v = (t < mink) ? scores[b * CDIM + t] : -__builtin_inff();
        red[t] = v; __syncthreads();
        for (int st = 128; st > 0; st >>= 1) {
            if (t < st) red[t] = fmaxf(red[t], red[t + st]);
            __syncthreads();
        }
        float M = red[0]; __syncthreads();
        float e = (t < mink) ? expf(v - M) : 0.0f;
        red[t] = e; __syncthreads();
        for (int st = 128; st > 0; st >>= 1) {
            if (t < st) red[t] += red[t + st];
            __syncthreads();
        }
        float S = red[0]; __syncthreads();
        if (t < mink) w[b * CDIM + t] = e / S;
    }
}

// ---------------------------------------------------------------------------
// Kernel 7: mask + apply: out = f_p * (1 + sum_k w_k * sigmoid(f_p[sel_k]))
// ---------------------------------------------------------------------------
__global__ __launch_bounds__(256)
void apply_mask_kernel(const float* __restrict__ fp,
                       const float* __restrict__ w,
                       const int* __restrict__ vals,
                       const int* __restrict__ minkp,
                       float* __restrict__ out)
{
    __shared__ float wsh[128];
    __shared__ int   ish[128];
    const int b     = blockIdx.x / 36;
    const int chunk = blockIdx.x % 36;
    const int t     = threadIdx.x;
    const int mink  = minkp[0];
    if (t < mink) {
        wsh[t] = w[b * CDIM + t];
        ish[t] = vals[b * CDIM + t];
    }
    __syncthreads();
    const int hw = chunk * 256 + t;
    const float* fpb = fp + (size_t)b * CDIM * HWDIM;
    float mask = 0.0f;
    for (int k = 0; k < mink; ++k) {
        float x = fpb[(size_t)ish[k] * HWDIM + hw];
        mask += wsh[k] / (1.0f + expf(-x));
    }
    const float m1 = 1.0f + mask;
    float* ob = out + (size_t)b * CDIM * HWDIM;
    for (int c = 0; c < CDIM; ++c) {
        size_t o = (size_t)c * HWDIM + hw;
        ob[o] = fpb[o] * m1;
    }
}

// ---------------------------------------------------------------------------
extern "C" void kernel_launch(void* const* d_in, const int* in_sizes, int n_in,
                              void* d_out, int out_size, void* d_ws, size_t ws_size,
                              hipStream_t stream)
{
    (void)in_sizes; (void)n_in; (void)out_size; (void)ws_size;

    const float* fp  = (const float*)d_in[0];   // (16,256,96,96) f32
    const float* fms = (const float*)d_in[1];   // (16,256,48,48) f32
    float* out = (float*)d_out;
    char*  ws  = (char*)d_ws;

    constexpr size_t nBC   = (size_t)BATCH * CDIM;          // 4096
    constexpr size_t nBCK  = nBC * KDIM;                    // 37,748,736

    const size_t off_msd    = 0;
    const size_t off_pd     = off_msd    + nBCK * 2;        // bf16
    const size_t off_sm     = off_pd     + nBCK * 2;        // bf16
    const size_t off_meanp  = off_sm     + nBC * CDIM * 4;  // f32 s_m
    const size_t off_pn     = off_meanp  + nBC * 4;
    const size_t off_msn    = off_pn     + nBC * 4;
    const size_t off_scores = off_msn    + nBC * 4;
    const size_t off_vals   = off_scores + nBC * 4;
    const size_t off_w      = off_vals   + nBC * 4;
    const size_t off_cnt    = off_w      + nBC * 4;
    const size_t off_mink   = off_cnt    + 64;

    __bf16* msd   = (__bf16*)(ws + off_msd);
    __bf16* pd    = (__bf16*)(ws + off_pd);
    float* sm     = (float*)(ws + off_sm);
    float* meanp  = (float*)(ws + off_meanp);
    float* pn     = (float*)(ws + off_pn);
    float* msn    = (float*)(ws + off_msn);
    float* scores = (float*)(ws + off_scores);
    int*   vals   = (int*)  (ws + off_vals);
    float* wbuf   = (float*)(ws + off_w);
    int*   cnt    = (int*)  (ws + off_cnt);
    int*   minkp  = (int*)  (ws + off_mink);

    stats_p_kernel   <<<nBC, 256, 0, stream>>>(fp, meanp, pn);
    center_p_kernel  <<<nBC, 256, 0, stream>>>(fp, meanp, pd);
    resize_ms_kernel <<<nBC, 256, 0, stream>>>(fms, msd, msn);
    gemm_score_kernel<<<BATCH * 16, 128, 0, stream>>>(msd, pd, msn, pn, sm);
    group_select_kernel<<<BATCH, 256, 0, stream>>>(sm, scores, vals, cnt);
    mink_weights_kernel<<<1, 256, 0, stream>>>(cnt, scores, minkp, wbuf);
    apply_mask_kernel<<<BATCH * 36, 256, 0, stream>>>(fp, wbuf, vals, minkp, out);
}
